// VLLMSSEGLA_73976516706664
// MI455X (gfx1250) — matrix operand order repaired and checked by reading, not compiled, not run on hardware
//
#include <hip/hip_runtime.h>
#include <cstdint>
#include <cstddef>

// ---------------- static problem dimensions (from reference) ----------------
#define H_      6
#define DK_     256
#define DV_     256
#define HID_    2048
#define KD_     1536          // H_*DK_
#define NEXP_   4
#define LOW_    16
#define L_      2048
#define NSAMP_  3             // cu_seqlens has 4 entries -> 3 packed samples
#define GATE_NORM_ 16.0f
#define EPS_    1e-5f

typedef __attribute__((ext_vector_type(16))) __bf16 v16bf;
typedef __attribute__((ext_vector_type(8)))  __bf16 v8bf;
typedef __attribute__((ext_vector_type(8)))  float  v8f;
typedef __attribute__((ext_vector_type(4)))  float  v4f;
typedef __attribute__((ext_vector_type(4)))  int    v4i;

// Async global->LDS (CDNA5 GLOBAL_LOAD_ASYNC_TO_LDS_B128, ASYNCcnt-tracked).
// Param types decoded from the round-2 diagnostic:
//   arg0: AS(1) int4*  (printed as "int __vector(4) __device__ *")
//   arg1: AS(3) int4*
#if defined(__has_builtin)
# if __has_builtin(__builtin_amdgcn_global_load_async_to_lds_b128) && \
     __has_builtin(__builtin_amdgcn_s_wait_asynccnt)
#  define USE_ASYNC_LDS 1
# endif
#endif
#ifndef USE_ASYNC_LDS
# define USE_ASYNC_LDS 0
#endif

#if USE_ASYNC_LDS
typedef __attribute__((address_space(1))) v4i as1_v4i;
typedef __attribute__((address_space(3))) v4i as3_v4i;
#endif

// ---------------------------------------------------------------------------
// f32 -> bf16 conversion (vectorized; all sizes here are multiples of 4)
// ---------------------------------------------------------------------------
__global__ void cvt_f32_to_bf16_kernel(const float* __restrict__ src,
                                       __bf16* __restrict__ dst, int n) {
  int i = (blockIdx.x * blockDim.x + threadIdx.x) * 4;
  if (i >= n) return;
  v4f v = *(const v4f*)(src + i);
  dst[i + 0] = (__bf16)v[0];
  dst[i + 1] = (__bf16)v[1];
  dst[i + 2] = (__bf16)v[2];
  dst[i + 3] = (__bf16)v[3];
}

// ---------------------------------------------------------------------------
// Tiled bf16 GEMM with v_wmma_f32_16x16x32_bf16.
// C[M,N] (f32) = A[M,K](bf16, row-major) * B[K,N](bf16, row-major)
// Block: 256 threads = 8 waves. Block tile 128x64, K-step 32.
// Wave w: wm = w&3 (32-row strip), wn = w>>2 (32-col strip); each wave owns a
// 32x32 output = 4 WMMA accumulators, reusing 2 A-frags x 2 B-frags per step
// (8 ds_load_b128 -> 4 WMMA).
// Fragment layouts per ISA 16-bit VGPR maps:
//   A 16x32 bf16: lane<16 -> row=lane,     K {0..7, 16..23}
//                 lane>=16 -> row=lane-16, K {8..15, 24..31}
//   B 32x16 bf16: lane<16 -> col=lane, K {0..7, 16..23} ; lane>=16 -> col-16, K {8..15, 24..31}
//   C 16x16 f32 : vgpr i, lane<16 -> (M=i, N=lane); lane>=16 -> (M=8+i, N=lane-16)
// A tile staged via async global->LDS when available; B staged transposed.
// ---------------------------------------------------------------------------
#define BM 128
#define BN 64
#define BK 32
#define AST 40   // padded LDS row stride in elements (80B, 16B aligned)
#define BST 40

__device__ __forceinline__ v16bf frag_gap(const __bf16* p, int gap) {
  v8bf lo = *(const v8bf*)p;
  v8bf hi = *(const v8bf*)(p + gap);
  v16bf f;
#pragma unroll
  for (int j = 0; j < 8; ++j) { f[j] = lo[j]; f[j + 8] = hi[j]; }
  return f;
}

__global__ __launch_bounds__(256) void gemm_bf16_wmma_kernel(
    const __bf16* __restrict__ A, const __bf16* __restrict__ B,
    float* __restrict__ C, int M, int N, int K) {
  __shared__ __align__(16) __bf16 As[BM * AST];  // 10 KB
  __shared__ __align__(16) __bf16 Bs[BN * BST];  //  5 KB (transposed: Bs[n][k])

  const int tid  = threadIdx.x;
  const int lane = tid & 31;
  const int wave = tid >> 5;
  const int wm   = wave & 3;    // which 32-row strip of the 128
  const int wn   = wave >> 2;   // which 32-col strip of the 64
  const int m0   = blockIdx.y * BM;
  const int n0   = blockIdx.x * BN;

  v8f acc00 = {}, acc01 = {}, acc10 = {}, acc11 = {};

  // staging assignments
  const int ar = tid >> 1;            // 0..127 : A tile row
  const int ac = (tid & 1) * 16;      // 0/16   : A col base (16 bf16 = 32B)
  const int bk = tid >> 3;            // 0..31  : B tile row (K)
  const int bc = (tid & 7) * 8;       // 0..56  : B tile col (N), 8 bf16

  const __bf16* aptr  = A + (size_t)(m0 + ar) * K + ac;
  const __bf16* bptr  = B + (size_t)bk * N + n0 + bc;
  __bf16*       asdst = &As[ar * AST + ac];

  for (int kt = 0; kt < K; kt += BK) {
    // ---- stage A tile: 32B per thread, contiguous in global AND LDS ----
#if USE_ASYNC_LDS
    // INST_OFFSET is added to BOTH global and LDS addresses (ISA 10.x), so two
    // b128 issues at offsets 0/16 cover the 32B per-thread chunk.
    __builtin_amdgcn_global_load_async_to_lds_b128(
        (as1_v4i*)(aptr + kt), (as3_v4i*)asdst, 0, 0);
    __builtin_amdgcn_global_load_async_to_lds_b128(
        (as1_v4i*)(aptr + kt), (as3_v4i*)asdst, 16, 0);
#else
    {
      v8bf a0v = *(const v8bf*)(aptr + kt);
      v8bf a1v = *(const v8bf*)(aptr + kt + 8);
      *(v8bf*)asdst = a0v;
      *(v8bf*)(asdst + 8) = a1v;
    }
#endif
    // ---- stage B tile transposed: Bs[n][k] ----
    v8bf bv = *(const v8bf*)(bptr + (size_t)kt * N);
#pragma unroll
    for (int j = 0; j < 8; ++j) Bs[(bc + j) * BST + bk] = bv[j];

    if (kt + BK < K) {  // global_prefetch_b8 of next tiles
      __builtin_prefetch(aptr + kt + BK, 0, 1);
      __builtin_prefetch(bptr + (size_t)(kt + BK) * N, 0, 1);
    }
#if USE_ASYNC_LDS
    __builtin_amdgcn_s_wait_asynccnt(0);
#endif
    __syncthreads();

    // ---- fragments ----
    const int lr  = lane & 15;
    const int kb  = (lane >> 4) * 8;    // A: 0 or 8
    const int bcc = lane & 15;
    const int bkb = (lane >> 4) * 8;    // B: 0 or 8
    v16bf a0 = frag_gap(&As[(wm * 32 +  0 + lr) * AST + kb], 16);
    v16bf a1 = frag_gap(&As[(wm * 32 + 16 + lr) * AST + kb], 16);
    v16bf b0 = frag_gap(&Bs[(wn * 32 +  0 + bcc) * BST + bkb], 16);
    v16bf b1 = frag_gap(&Bs[(wn * 32 + 16 + bcc) * BST + bkb], 16);

    acc00 = __builtin_amdgcn_wmma_f32_16x16x32_bf16(false, a0, false, b0,
                                                    (short)0, acc00, false, false);
    acc01 = __builtin_amdgcn_wmma_f32_16x16x32_bf16(false, a0, false, b1,
                                                    (short)0, acc01, false, false);
    acc10 = __builtin_amdgcn_wmma_f32_16x16x32_bf16(false, a1, false, b0,
                                                    (short)0, acc10, false, false);
    acc11 = __builtin_amdgcn_wmma_f32_16x16x32_bf16(false, a1, false, b1,
                                                    (short)0, acc11, false, false);
    __syncthreads();
  }

  // ---- store C per 16x16 f32 layout ----
  const int cn = lane & 15;
  const int mo = (lane >> 4) * 8;
  float* crow0 = C + (size_t)(m0 + wm * 32 + mo) * N + n0 + wn * 32 + cn;       // msub 0
  float* crow1 = C + (size_t)(m0 + wm * 32 + 16 + mo) * N + n0 + wn * 32 + cn;  // msub 1
#pragma unroll
  for (int i = 0; i < 8; ++i) {
    crow0[(size_t)i * N]      = acc00[i];
    crow0[(size_t)i * N + 16] = acc01[i];
    crow1[(size_t)i * N]      = acc10[i];
    crow1[(size_t)i * N + 16] = acc11[i];
  }
}

// ---------------------------------------------------------------------------
// Small projections (N in {4,16}): y[L,Ns] = x[L,HID] @ W[HID,Ns]
// ---------------------------------------------------------------------------
__global__ void small_proj_kernel(const float* __restrict__ x,
                                  const float* __restrict__ W,
                                  float* __restrict__ y, int Ns) {
  int t = blockIdx.x;
  int j = threadIdx.x;
  if (j >= Ns) return;
  const float* xr = x + (size_t)t * HID_;
  float acc = 0.f;
  for (int kx = 0; kx < HID_; ++kx) acc = fmaf(xr[kx], W[kx * Ns + j], acc);
  y[t * Ns + j] = acc;
}

// gk = log_sigmoid(t16 @ W1 + b) / GATE_NORM   (K = LOW_ = 16)
__global__ void gate_stage2_kernel(const float* __restrict__ t16,
                                   const float* __restrict__ W1,
                                   const float* __restrict__ b,
                                   float* __restrict__ gk) {
  int idx = blockIdx.x * blockDim.x + threadIdx.x;
  if (idx >= L_ * KD_) return;
  int t = idx / KD_, n = idx % KD_;
  const float* tr = t16 + t * LOW_;
  float acc = b[n];
#pragma unroll
  for (int kx = 0; kx < LOW_; ++kx) acc = fmaf(tr[kx], W1[kx * KD_ + n], acc);
  float ls = fminf(acc, 0.f) - log1pf(expf(-fabsf(acc)));  // stable log_sigmoid
  gk[idx] = ls / GATE_NORM_;
}

// in-place softmax over the DK=256 feature dim, one block per (t,h) row
__global__ void softmax_head_kernel(float* __restrict__ k) {
  int row = blockIdx.x;
  int t = row / H_, h = row % H_;
  float* p = k + (size_t)t * KD_ + h * DK_;
  int d = threadIdx.x;
  __shared__ float red[256];
  float xv = p[d];
  red[d] = xv; __syncthreads();
  for (int s = 128; s > 0; s >>= 1) {
    if (d < s) red[d] = fmaxf(red[d], red[d + s]);
    __syncthreads();
  }
  float m = red[0]; __syncthreads();
  float e = expf(xv - m);
  red[d] = e; __syncthreads();
  for (int s = 128; s > 0; s >>= 1) {
    if (d < s) red[d] += red[d + s];
    __syncthreads();
  }
  p[d] = e / red[0];
}

// top-1 routing: expert = argmax(eta), val = softmax(eta)[argmax] = 1/sum(exp(e-m))
__global__ void route_kernel(const float* __restrict__ eta,
                             int* __restrict__ expert, float* __restrict__ val) {
  int t = blockIdx.x * blockDim.x + threadIdx.x;
  if (t >= L_) return;
  float e[NEXP_];
  float m = -1e30f; int best = 0;
#pragma unroll
  for (int i = 0; i < NEXP_; ++i) {
    e[i] = eta[t * NEXP_ + i];
    if (e[i] > m) { m = e[i]; best = i; }
  }
  float s = 0.f;
#pragma unroll
  for (int i = 0; i < NEXP_; ++i) s += expf(e[i] - m);
  expert[t] = best;
  val[t] = 1.f / s;
}

// stable bucket-by-(sample,expert), preserving time order == the reference's
// bit-packed argsort for TOPK=1. O(L) serial, single thread (L=2048).
__global__ void sort_kernel(const int* __restrict__ cu,
                            const int* __restrict__ expert,
                            const float* __restrict__ val,
                            int* __restrict__ gsort, float* __restrict__ vsort,
                            int* __restrict__ segstart) {
  if (threadIdx.x != 0 || blockIdx.x != 0) return;
  int counts[NSAMP_ * NEXP_];
  for (int i = 0; i < NSAMP_ * NEXP_; ++i) counts[i] = 0;
  for (int t = 0; t < L_; ++t) {
    int s = 0;
    while (s + 1 < NSAMP_ && t >= cu[s + 1]) ++s;
    counts[s * NEXP_ + expert[t]]++;
  }
  int off = 0;
  for (int i = 0; i < NSAMP_ * NEXP_; ++i) { segstart[i] = off; off += counts[i]; counts[i] = 0; }
  segstart[NSAMP_ * NEXP_] = off;  // == L_
  for (int t = 0; t < L_; ++t) {
    int s = 0;
    while (s + 1 < NSAMP_ && t >= cu[s + 1]) ++s;
    int seg = s * NEXP_ + expert[t];
    int p = segstart[seg] + counts[seg]++;
    gsort[p] = t;
    vsort[p] = val[t];
  }
}

// ---------------------------------------------------------------------------
// GLA recurrence, one workgroup per (segment, head). State S[256,256] f32 is
// held in registers: 1024 threads x 64 elements. Thread tid owns column
// dv = tid&255 for dk in [ (tid>>8)*64, +64 ). Per step the 256 q/k/exp(gk)/v
// values are staged in LDS and broadcast-read; output reduced 4-way in LDS.
// ---------------------------------------------------------------------------
__global__ __launch_bounds__(1024) void gla_scan_kernel(
    const float* __restrict__ q, const float* __restrict__ k,
    const float* __restrict__ v, const float* __restrict__ gk,
    float* __restrict__ o, const int* __restrict__ segstart,
    const int* __restrict__ gather, const float* __restrict__ scale,
    int accumulate) {
  const int seg = blockIdx.x;
  const int h   = blockIdx.y;
  const int p0 = segstart[seg], p1 = segstart[seg + 1];
  if (p1 <= p0) return;

  const int tid = threadIdx.x;
  const int dv  = tid & (DV_ - 1);
  const int dkg = tid >> 8;        // 0..3
  const int dk0 = dkg * 64;

  __shared__ float qs[DK_], ks[DK_], es[DK_], vs[DV_];
  __shared__ float partial[1024];

  float S[64];
#pragma unroll
  for (int i = 0; i < 64; ++i) S[i] = 0.f;

  for (int p = p0; p < p1; ++p) {
    const int idx = gather ? gather[p] : p;
    const float sc = scale ? scale[p] : 1.f;
    if (tid < DK_) {
      size_t base = (size_t)idx * KD_ + (size_t)h * DK_;
      qs[tid] = q[base + tid] * sc;   // emulq
      ks[tid] = k[base + tid] * sc;   // emulk
      es[tid] = __expf(gk[base + tid]);
      vs[tid] = v[base + tid];
    }
    __syncthreads();

    const float vv = vs[dv];
    float acc = 0.f;
#pragma unroll
    for (int i = 0; i < 64; ++i) {
      const int dk = dk0 + i;
      float s2 = es[dk] * S[i] + ks[dk] * vv;  // S = e*S + k v^T
      S[i] = s2;
      acc = fmaf(qs[dk], s2, acc);             // o += q^T S (partial over dk)
    }
    partial[tid] = acc;
    __syncthreads();
    if (dkg == 0) {
      float out = partial[dv] + partial[DV_ + dv] +
                  partial[2 * DV_ + dv] + partial[3 * DV_ + dv];
      size_t ob = (size_t)idx * KD_ + (size_t)h * DV_ + dv;
      if (accumulate) o[ob] += out; else o[ob] = out;
    }
    __syncthreads();
  }
}

// gated RMSNorm -> bf16 (feeds the final WMMA GEMM with Wo)
__global__ void rmsnorm_gate_kernel(const float* __restrict__ o,
                                    const float* __restrict__ g,
                                    const float* __restrict__ w,
                                    __bf16* __restrict__ onb) {
  int row = blockIdx.x;
  int t = row / H_, h = row % H_;
  int d = threadIdx.x;
  size_t base = (size_t)t * KD_ + (size_t)h * DV_;
  float ov = o[base + d];
  __shared__ float red[256];
  red[d] = ov * ov; __syncthreads();
  for (int s = 128; s > 0; s >>= 1) {
    if (d < s) red[d] += red[d + s];
    __syncthreads();
  }
  float rms = sqrtf(red[0] / (float)DV_ + EPS_);
  float gv = g[base + d];
  float silu = gv / (1.f + expf(-gv));
  onb[base + d] = (__bf16)((ov / rms) * w[d] * silu);
}

// ---------------------------------------------------------------------------
// host-side orchestration
// ---------------------------------------------------------------------------
static inline int cdiv(int a, int b) { return (a + b - 1) / b; }

extern "C" void kernel_launch(void* const* d_in, const int* in_sizes, int n_in,
                              void* d_out, int out_size, void* d_ws, size_t ws_size,
                              hipStream_t stream) {
  (void)in_sizes; (void)n_in; (void)out_size; (void)ws_size;
  const float* x      = (const float*)d_in[0];
  const int*   cu     = (const int*)  d_in[1];
  const float* Wq     = (const float*)d_in[2];
  const float* Wk     = (const float*)d_in[3];
  const float* Wv     = (const float*)d_in[4];
  const float* Wlq0   = (const float*)d_in[5];
  const float* Wlq1   = (const float*)d_in[6];
  const float* Wlk0   = (const float*)d_in[7];
  const float* Wlk1   = (const float*)d_in[8];
  const float* Wgk00  = (const float*)d_in[9];
  const float* Wgk01  = (const float*)d_in[10];
  const float* bgk01  = (const float*)d_in[11];
  const float* Wgk10  = (const float*)d_in[12];
  const float* Wgk11  = (const float*)d_in[13];
  const float* bgk11  = (const float*)d_in[14];
  const float* We     = (const float*)d_in[15];
  const float* Wg0    = (const float*)d_in[16];
  const float* Wg1    = (const float*)d_in[17];
  const float* onormw = (const float*)d_in[18];
  const float* Wo     = (const float*)d_in[19];
  float* out = (float*)d_out;

  // -------- workspace carve-out --------
  char* wsp = (char*)d_ws;
  auto alloc = [&](size_t bytes) -> char* {
    char* p = wsp;
    wsp += (bytes + 255) & ~(size_t)255;
    return p;
  };
  __bf16* xb     = (__bf16*)alloc((size_t)L_ * HID_ * 2);
  __bf16* Wq_b   = (__bf16*)alloc((size_t)HID_ * KD_ * 2);
  __bf16* Wk_b   = (__bf16*)alloc((size_t)HID_ * KD_ * 2);
  __bf16* Wv_b   = (__bf16*)alloc((size_t)HID_ * KD_ * 2);
  __bf16* Wlq0_b = (__bf16*)alloc((size_t)HID_ * DV_ * 2);
  __bf16* Wlq1_b = (__bf16*)alloc((size_t)DV_ * KD_ * 2);
  __bf16* Wlk0_b = (__bf16*)alloc((size_t)HID_ * DV_ * 2);
  __bf16* Wlk1_b = (__bf16*)alloc((size_t)DV_ * KD_ * 2);
  __bf16* Wg0_b  = (__bf16*)alloc((size_t)HID_ * DV_ * 2);
  __bf16* Wg1_b  = (__bf16*)alloc((size_t)DV_ * KD_ * 2);
  __bf16* Wo_b   = (__bf16*)alloc((size_t)KD_ * HID_ * 2);
  float*  q1   = (float*)alloc((size_t)L_ * KD_ * 4);
  float*  k1   = (float*)alloc((size_t)L_ * KD_ * 4);
  float*  vb   = (float*)alloc((size_t)L_ * KD_ * 4);
  float*  q2   = (float*)alloc((size_t)L_ * KD_ * 4);
  float*  k2   = (float*)alloc((size_t)L_ * KD_ * 4);
  float*  gk1  = (float*)alloc((size_t)L_ * KD_ * 4);
  float*  gk2  = (float*)alloc((size_t)L_ * KD_ * 4);
  float*  ob   = (float*)alloc((size_t)L_ * KD_ * 4);
  float*  gb   = (float*)alloc((size_t)L_ * KD_ * 4);
  float*  t256 = (float*)alloc((size_t)L_ * DV_ * 4);
  __bf16* t256b = (__bf16*)alloc((size_t)L_ * DV_ * 2);
  float*  t16  = (float*)alloc((size_t)L_ * LOW_ * 4);
  float*  eta  = (float*)alloc((size_t)L_ * NEXP_ * 4);
  int*    expert = (int*)alloc((size_t)L_ * 4);
  float*  val    = (float*)alloc((size_t)L_ * 4);
  int*    gsort  = (int*)alloc((size_t)L_ * 4);
  float*  vsort  = (float*)alloc((size_t)L_ * 4);
  int*    segst  = (int*)alloc((NSAMP_ * NEXP_ + 1) * 4);
  __bf16* onb  = (__bf16*)alloc((size_t)L_ * KD_ * 2);

  auto cvt = [&](const float* s, __bf16* d, int n) {
    cvt_f32_to_bf16_kernel<<<cdiv(n / 4, 256), 256, 0, stream>>>(s, d, n);
  };
  auto gemm = [&](const __bf16* A, const __bf16* B, float* C, int M, int N, int K) {
    gemm_bf16_wmma_kernel<<<dim3(N / BN, M / BM), 256, 0, stream>>>(A, B, C, M, N, K);
  };

  // -------- bf16 conversions --------
  cvt(x, xb, L_ * HID_);
  cvt(Wq, Wq_b, HID_ * KD_);
  cvt(Wk, Wk_b, HID_ * KD_);
  cvt(Wv, Wv_b, HID_ * KD_);
  cvt(Wlq0, Wlq0_b, HID_ * DV_);
  cvt(Wlq1, Wlq1_b, DV_ * KD_);
  cvt(Wlk0, Wlk0_b, HID_ * DV_);
  cvt(Wlk1, Wlk1_b, DV_ * KD_);
  cvt(Wg0, Wg0_b, HID_ * DV_);
  cvt(Wg1, Wg1_b, DV_ * KD_);
  cvt(Wo, Wo_b, KD_ * HID_);

  // -------- WMMA projection GEMMs --------
  gemm(xb, Wq_b, q1, L_, KD_, HID_);
  gemm(xb, Wk_b, k1, L_, KD_, HID_);
  gemm(xb, Wv_b, vb, L_, KD_, HID_);
  // low-rank q2
  gemm(xb, Wlq0_b, t256, L_, DV_, HID_);
  cvt(t256, t256b, L_ * DV_);
  gemm(t256b, Wlq1_b, q2, L_, KD_, DV_);
  // low-rank k2
  gemm(xb, Wlk0_b, t256, L_, DV_, HID_);
  cvt(t256, t256b, L_ * DV_);
  gemm(t256b, Wlk1_b, k2, L_, KD_, DV_);
  // gate g
  gemm(xb, Wg0_b, t256, L_, DV_, HID_);
  cvt(t256, t256b, L_ * DV_);
  gemm(t256b, Wg1_b, gb, L_, KD_, DV_);

  // -------- gate projections (small) --------
  small_proj_kernel<<<L_, LOW_, 0, stream>>>(x, Wgk00, t16, LOW_);
  gate_stage2_kernel<<<cdiv(L_ * KD_, 256), 256, 0, stream>>>(t16, Wgk01, bgk01, gk1);
  small_proj_kernel<<<L_, LOW_, 0, stream>>>(x, Wgk10, t16, LOW_);
  gate_stage2_kernel<<<cdiv(L_ * KD_, 256), 256, 0, stream>>>(t16, Wgk11, bgk11, gk2);

  // -------- routing --------
  small_proj_kernel<<<L_, NEXP_, 0, stream>>>(x, We, eta, NEXP_);
  route_kernel<<<cdiv(L_, 256), 256, 0, stream>>>(eta, expert, val);
  sort_kernel<<<1, 32, 0, stream>>>(cu, expert, val, gsort, vsort, segst);

  // -------- key feature-map softmax --------
  softmax_head_kernel<<<L_ * H_, 256, 0, stream>>>(k1);
  softmax_head_kernel<<<L_ * H_, 256, 0, stream>>>(k2);

  // -------- GLA scans: base path writes, sparse path accumulates --------
  gla_scan_kernel<<<dim3(NSAMP_, H_), 1024, 0, stream>>>(
      q1, k1, vb, gk1, ob, cu, nullptr, nullptr, 0);
  gla_scan_kernel<<<dim3(NSAMP_ * NEXP_, H_), 1024, 0, stream>>>(
      q2, k2, vb, gk2, ob, segst, gsort, vsort, 1);

  // -------- gated RMSNorm -> bf16, final WMMA GEMM with Wo --------
  rmsnorm_gate_kernel<<<L_ * H_, 256, 0, stream>>>(ob, gb, onormw, onb);
  gemm(onb, Wo_b, out, L_, HID_, KD_);
}
